// CombineGraph_9509057593869
// MI455X (gfx1250) — matrix operand order repaired and checked by reading, not compiled, not run on hardware
//
#include <hip/hip_runtime.h>
#include <hip/hip_bf16.h>

typedef __attribute__((ext_vector_type(16))) _Float16 v16h;
typedef __attribute__((ext_vector_type(8)))  float    v8f;

#define NNODE 40000
#define DIM   100
#define BB    128
#define NN    70
#define EE    640000

// padded dims for WMMA tiling
#define NP    80     // padded N (M rows / j cols), 5 tiles of 16
#define KD    128    // padded D as K for GEMM1 (4 chunks of 32)
#define KJ    96     // padded N as K for GEMM2 (3 chunks of 32)
#define DP    112    // padded D as N-tiles for GEMM2 (7 tiles of 16)

// ---- dynamic LDS layout for intra kernel ----
#define OFF_ROWS   0                       // 80 ints        ->   320 B
#define OFF_HH     (OFF_ROWS + 320)        // [80][128] f16  -> 20480 B
#define OFF_HAH    (OFF_HH + 20480)        // [80][128] f16  -> 20480 B
#define OFF_HT     (OFF_HAH + 20480)       // [112][96] f16  -> 21504 B
#define OFF_ALPHA  (OFF_HT + 21504)        // [80][80] f32   -> 25600 B
#define OFF_ALPHAH (OFF_ALPHA + 25600)     // [80][96] f16   -> 15360 B
#define SMEM_INTRA (OFF_ALPHAH + 15360)    // = 103744 B  (< 320 KB WGP LDS)

// ---- dynamic LDS layout for session kernel ----
#define S_OFF_DA   0                       // [128][128] f32 -> 65536 B
#define S_OFF_S    (S_OFF_DA + 65536)      // [128][100] f32 -> 51200 B
#define S_OFF_S2   (S_OFF_S + 51200)
#define S_OFF_ACC  (S_OFF_S2 + 51200)
#define SMEM_SESS  (S_OFF_ACC + 51200)     // = 219136 B

// A fragment (16-bit, 16x32 MxK): lane<16 holds row m0+lane, K = k0+0..7 and k0+16..23;
// lane>=16 holds row m0+lane-16, K = k0+8..15 and k0+24..31.
__device__ __forceinline__ v16h load_a16(const _Float16* mat, int ldm, int m0, int k0, int lane) {
  int m  = m0 + (lane & 15);
  int kb = k0 + ((lane & 16) ? 8 : 0);
  const _Float16* p = mat + m * ldm + kb;
  v16h r;
#pragma unroll
  for (int t = 0; t < 8; ++t) { r[t] = p[t]; r[t + 8] = p[t + 16]; }
  return r;
}

// B fragment (16-bit, 32x16 KxN): lane<16 holds col n0+lane with K = k0+0..7 and k0+16..23;
// lane>=16 holds col n0+lane-16 with K = k0+8..15 and k0+24..31.
// `mat` stores B column-contiguous-in-K: mat[n][k] row-major with leading dim ldm.
__device__ __forceinline__ v16h load_b16(const _Float16* mat, int ldm, int n0, int k0, int lane) {
  int n  = n0 + (lane & 15);
  int kb = k0 + ((lane & 16) ? 8 : 0);
  const _Float16* p = mat + n * ldm + kb;
  v16h r;
#pragma unroll
  for (int t = 0; t < 8; ++t) { r[t] = p[t]; r[t + 8] = p[t + 16]; }
  return r;
}

// ---------------- intra-session GAT kernel (WMMA) : one block per batch ----------------
extern "C" __global__ void intra_kernel(const int* __restrict__ inputs,
                                        const int* __restrict__ edge,
                                        const float* __restrict__ emb,
                                        const float* __restrict__ a0,
                                        const float* __restrict__ a1,
                                        const float* __restrict__ a2,
                                        const float* __restrict__ a3,
                                        float* __restrict__ out) {
  extern __shared__ char smem[];
  int*       rowsIdx = (int*)(smem + OFF_ROWS);
  _Float16*  hH      = (_Float16*)(smem + OFF_HH);
  _Float16*  haH     = (_Float16*)(smem + OFF_HAH);
  _Float16*  hT      = (_Float16*)(smem + OFF_HT);
  float*     alpha   = (float*)(smem + OFF_ALPHA);
  _Float16*  alphaH  = (_Float16*)(smem + OFF_ALPHAH);

  const int bb   = blockIdx.x;
  const int tid  = threadIdx.x;
  const int w    = tid >> 5;     // wave id (wave32)
  const int lane = tid & 31;

  if (tid < NN) rowsIdx[tid] = inputs[bb * NN + tid];
  __syncthreads();

  // stage h (f16, zero padded) and hT (transposed, f16, zero padded)
  for (int idx = tid; idx < NP * KD; idx += 256) {
    int i = idx >> 7, d = idx & (KD - 1);
    float v = 0.f;
    if (i < NN && d < DIM) v = emb[(size_t)rowsIdx[i] * DIM + d];
    hH[idx] = (_Float16)v;
  }
  for (int idx = tid; idx < DP * KJ; idx += 256) {
    int d = idx / KJ, j = idx - d * KJ;
    float v = 0.f;
    if (j < NN && d < DIM) v = emb[(size_t)rowsIdx[j] * DIM + d];
    hT[idx] = (_Float16)v;
  }
  for (int idx = tid; idx < NP * NP; idx += 256) alpha[idx] = -9.0e15f;
  __syncthreads();

  // ---- GEMM1 x4: e_k = leaky((h*a_k) @ h^T), scatter where edge==k+1 ----
  for (int k = 0; k < 4; ++k) {
    const float* av = (k == 0) ? a0 : ((k == 1) ? a1 : ((k == 2) ? a2 : a3));
    for (int idx = tid; idx < NP * KD; idx += 256) {
      int i = idx >> 7, d = idx & (KD - 1);
      float v = 0.f;
      if (i < NN && d < DIM) v = emb[(size_t)rowsIdx[i] * DIM + d] * av[d];
      haH[idx] = (_Float16)v;
    }
    __syncthreads();

    for (int t = w; t < 25; t += 8) {
      int i0 = (t / 5) * 16, j0 = (t % 5) * 16;
      v8f c = {};
#pragma unroll
      for (int kk = 0; kk < KD / 32; ++kk) {
        v16h af = load_a16(haH, KD, i0, kk * 32, lane);
        v16h bf = load_b16(hH,  KD, j0, kk * 32, lane);
        c = __builtin_amdgcn_wmma_f32_16x16x32_f16(false, af, false, bf,
                                                   (short)0, c, false, false);
      }
      // C layout: VGPR r -> (M=i0+r | i0+8+r, N=j0 + lane%16)
      int iBase = i0 + ((lane & 16) ? 8 : 0);
      int j = j0 + (lane & 15);
#pragma unroll
      for (int r = 0; r < 8; ++r) {
        int i = iBase + r;
        if (i < NN && j < NN) {
          float v = c[r];
          v = (v >= 0.f) ? v : 0.2f * v;      // leaky relu, alpha=0.2
          int e = edge[((size_t)bb * NN + i) * NN + j];
          if (e == k + 1) alpha[i * NP + j] = v;
        }
      }
    }
    __syncthreads();
  }

  // ---- row softmax over the 70 real columns; emit f16 alpha, K-padded to 96 ----
  for (int i = w; i < NP; i += 8) {
    if (i < NN) {
      int base = i * NP;
      float v1 = alpha[base + lane];
      float v2 = alpha[base + lane + 32];
      float v3 = (lane < 6) ? alpha[base + lane + 64] : -3.0e38f;
      float m = fmaxf(v1, fmaxf(v2, v3));
      for (int off = 16; off > 0; off >>= 1) m = fmaxf(m, __shfl_xor(m, off, 32));
      float p1 = __expf(v1 - m);
      float p2 = __expf(v2 - m);
      float p3 = (lane < 6) ? __expf(v3 - m) : 0.f;
      float s = p1 + p2 + p3;
      for (int off = 16; off > 0; off >>= 1) s += __shfl_xor(s, off, 32);
      float inv = 1.f / s;
      alphaH[i * KJ + lane]      = (_Float16)(p1 * inv);
      alphaH[i * KJ + lane + 32] = (_Float16)(p2 * inv);
      alphaH[i * KJ + lane + 64] = (lane < 6) ? (_Float16)(p3 * inv) : (_Float16)0.f;
    } else {
      alphaH[i * KJ + lane]      = (_Float16)0.f;
      alphaH[i * KJ + lane + 32] = (_Float16)0.f;
      alphaH[i * KJ + lane + 64] = (_Float16)0.f;
    }
  }
  __syncthreads();

  // ---- GEMM2: out = alpha @ h  (B read from pre-transposed hT) ----
  for (int t = w; t < 35; t += 8) {
    int i0 = (t / 7) * 16, d0 = (t % 7) * 16;
    v8f c = {};
#pragma unroll
    for (int kk = 0; kk < KJ / 32; ++kk) {
      v16h af = load_a16(alphaH, KJ, i0, kk * 32, lane);
      v16h bf = load_b16(hT,     KJ, d0, kk * 32, lane);
      c = __builtin_amdgcn_wmma_f32_16x16x32_f16(false, af, false, bf,
                                                 (short)0, c, false, false);
    }
    int iBase = i0 + ((lane & 16) ? 8 : 0);
    int d = d0 + (lane & 15);
#pragma unroll
    for (int r = 0; r < 8; ++r) {
      int i = iBase + r;
      if (i < NN && d < DIM) out[((size_t)bb * NN + i) * DIM + d] = c[r];
    }
  }
}

// ---------------- zero workspace ----------------
extern "C" __global__ void zero_kernel(float* __restrict__ p, long n) {
  long i = (long)blockIdx.x * blockDim.x + threadIdx.x;
  if (i < n) p[i] = 0.f;
}

// ---------------- one SpMM layer: y[rows[e]] += vals[e] * x[cols[e]]  (one wave per edge) ----------------
extern "C" __global__ void spmm_kernel(const float* __restrict__ x,
                                       const int* __restrict__ rows,
                                       const int* __restrict__ cols,
                                       const float* __restrict__ vals,
                                       float* __restrict__ y) {
  int gw   = (int)((blockIdx.x * blockDim.x + threadIdx.x) >> 5);
  int lane = threadIdx.x & 31;
  if (gw >= EE) return;
  int r = rows[gw], c = cols[gw];
  float v = vals[gw];
  if (lane < 25) {                       // 25 float4 = 100 floats per row
    const float4* xr = (const float4*)(x + (size_t)c * DIM);
    float4 xx = xr[lane];
    float* yp = y + (size_t)r * DIM + lane * 4;
    atomicAdd(yp + 0, v * xx.x);
    atomicAdd(yp + 1, v * xx.y);
    atomicAdd(yp + 2, v * xx.z);
    atomicAdd(yp + 3, v * xx.w);
  }
}

// ---------------- out += gather((emb+x1+x2)/3, sess_item with zero-pad row) ----------------
extern "C" __global__ void combine_kernel(const int* __restrict__ sess_item,
                                          const float* __restrict__ emb,
                                          const float* __restrict__ x1,
                                          const float* __restrict__ x2,
                                          float* __restrict__ out) {
  int idx = blockIdx.x * blockDim.x + threadIdx.x;
  if (idx >= BB * NN * DIM) return;
  int d  = idx % DIM;
  int bi = idx / DIM;                    // b*NN + i
  int item = sess_item[bi];
  float add = 0.f;
  if (item > 0) {
    size_t o = (size_t)(item - 1) * DIM + d;
    add = (emb[o] + x1[o] + x2[o]) * (1.f / 3.f);
  }
  out[idx] += add;
}

// ---------------- session embedding: s; DA=D@A; acc = (s + DA s + DA^2 s)/3 ----------------
extern "C" __global__ void session_kernel(const int* __restrict__ sess_item,
                                          const float* __restrict__ Dm,
                                          const float* __restrict__ Am,
                                          const float* __restrict__ sess_len,
                                          const float* __restrict__ emb,
                                          float* __restrict__ out_sess) {
  extern __shared__ char smem[];
  float* DA  = (float*)(smem + S_OFF_DA);
  float* s   = (float*)(smem + S_OFF_S);
  float* s2  = (float*)(smem + S_OFF_S2);
  float* acc = (float*)(smem + S_OFF_ACC);
  int tid = threadIdx.x;

  for (int idx = tid; idx < 128 * 128; idx += 256) {
    int r = idx >> 7, c = idx & 127;
    float sum = 0.f;
    for (int k = 0; k < 128; ++k) sum += Dm[r * 128 + k] * Am[k * 128 + c];
    DA[idx] = sum;
  }
  for (int idx = tid; idx < 128 * DIM; idx += 256) {
    int bq = idx / DIM, d = idx - bq * DIM;
    float sum = 0.f;
    for (int i = 0; i < NN; ++i) {
      int item = sess_item[bq * NN + i];
      if (item > 0) sum += emb[(size_t)(item - 1) * DIM + d];
    }
    float v = sum / sess_len[bq];
    s[idx] = v; acc[idx] = v;
  }
  __syncthreads();
  for (int L = 0; L < 2; ++L) {
    for (int idx = tid; idx < 128 * DIM; idx += 256) {
      int bq = idx / DIM, d = idx - bq * DIM;
      float sum = 0.f;
      for (int k = 0; k < 128; ++k) sum += DA[bq * 128 + k] * s[k * DIM + d];
      s2[idx] = sum;
    }
    __syncthreads();
    for (int idx = tid; idx < 128 * DIM; idx += 256) { float v = s2[idx]; s[idx] = v; acc[idx] += v; }
    __syncthreads();
  }
  for (int idx = tid; idx < 128 * DIM; idx += 256) out_sess[idx] = acc[idx] * (1.f / 3.f);
}

extern "C" void kernel_launch(void* const* d_in, const int* in_sizes, int n_in,
                              void* d_out, int out_size, void* d_ws, size_t ws_size,
                              hipStream_t stream) {
  (void)in_sizes; (void)n_in; (void)out_size; (void)ws_size;
  const int*   inputs    = (const int*)d_in[0];
  const int*   edge      = (const int*)d_in[1];
  const int*   sess_item = (const int*)d_in[4];
  const float* Dm        = (const float*)d_in[5];
  const float* Am        = (const float*)d_in[6];
  const float* sess_len  = (const float*)d_in[7];
  const float* emb       = (const float*)d_in[8];
  const float* a0        = (const float*)d_in[9];
  const float* a1        = (const float*)d_in[10];
  const float* a2        = (const float*)d_in[11];
  const float* a3        = (const float*)d_in[12];
  const int*   arows     = (const int*)d_in[13];
  const int*   acols     = (const int*)d_in[14];
  const float* avals     = (const float*)d_in[15];
  float* out = (float*)d_out;

  float* x1 = (float*)d_ws;                    // 40000*100 f32 = 16 MB
  float* x2 = x1 + (size_t)NNODE * DIM;        // 16 MB

  // 1) zero layer buffers (x1,x2 contiguous)
  long nzero = 2L * NNODE * DIM;
  zero_kernel<<<(int)((nzero + 255) / 256), 256, 0, stream>>>(x1, nzero);

  // 2) inter: two SpMM propagation layers (L2-resident gather + f32 atomics)
  int spmm_blocks = (EE + 7) / 8;              // 8 waves per 256-thread block
  spmm_kernel<<<spmm_blocks, 256, 0, stream>>>(emb, arows, acols, avals, x1);
  spmm_kernel<<<spmm_blocks, 256, 0, stream>>>(x1,  arows, acols, avals, x2);

  // 3) intra GAT via WMMA (writes intra result into d_out)
  intra_kernel<<<BB, 256, SMEM_INTRA, stream>>>(inputs, edge, emb, a0, a1, a2, a3, out);

  // 4) add gathered inter embeddings: out += inter_pad[sess_item]
  combine_kernel<<<(BB * NN * DIM + 255) / 256, 256, 0, stream>>>(sess_item, emb, x1, x2, out);

  // 5) session embedding tail of d_out
  session_kernel<<<1, 256, SMEM_SESS, stream>>>(sess_item, Dm, Am, sess_len, emb,
                                                out + (size_t)BB * NN * DIM);
}
